// CopyGenerator_67216238182745
// MI455X (gfx1250) — compile-verified, matrix operand order checked by reading, not run on hardware
//
#include <hip/hip_runtime.h>
#include <hip/hip_bf16.h>

// Problem dimensions (fixed by the reference)
#define BB   32
#define TT   128
#define BT   (BB*TT)      // 4096 rows
#define SS   400
#define VV   32000
#define CVV  600
#define DD   512
#define LDOUT (VV + CVV)  // 32600
#define PAD  1

typedef __attribute__((ext_vector_type(16))) __bf16 v16bf;
typedef __attribute__((ext_vector_type(8)))  __bf16 v8bf;
typedef __attribute__((ext_vector_type(8)))  float  v8f;
typedef __attribute__((ext_vector_type(4)))  int    v4i;

#if __has_builtin(__builtin_amdgcn_global_load_async_to_lds_b128)
#define USE_ASYNC_LDS 1
typedef __attribute__((address_space(1))) v4i* gptr_b128;  // global v4i*
typedef __attribute__((address_space(3))) v4i* lptr_b128;  // LDS v4i*
#else
#define USE_ASYNC_LDS 0
#endif

// ---------------------------------------------------------------------------
// fp32 -> bf16 (round to nearest even), stored as raw u16
// ---------------------------------------------------------------------------
__device__ __forceinline__ unsigned short f32_to_bf16(float f) {
  unsigned u = __float_as_uint(f);
  unsigned r = 0x7FFFu + ((u >> 16) & 1u);
  return (unsigned short)((u + r) >> 16);
}

__global__ __launch_bounds__(256) void cvt_bf16_kernel(const float* __restrict__ src,
                                                       unsigned short* __restrict__ dst,
                                                       int n) {
  int i = blockIdx.x * 256 + threadIdx.x;
  if (i < n) dst[i] = f32_to_bf16(src[i]);
}

// ---------------------------------------------------------------------------
// copy gate: gate[row] = sigmoid(hidden[row] . w_copy + b_copy)
// ---------------------------------------------------------------------------
__global__ __launch_bounds__(256) void gate_kernel(const float* __restrict__ hidden,
                                                   const float* __restrict__ w_copy,
                                                   const float* __restrict__ b_copy,
                                                   float* __restrict__ gate) {
  int row = blockIdx.x;
  int tid = threadIdx.x;
  const float* h = hidden + (size_t)row * DD;
  float acc = h[tid] * w_copy[tid] + h[tid + 256] * w_copy[tid + 256];
  __shared__ float red[256];
  red[tid] = acc;
  __syncthreads();
  for (int off = 128; off > 0; off >>= 1) {
    if (tid < off) red[tid] += red[tid + off];
    __syncthreads();
  }
  if (tid == 0) gate[row] = 1.0f / (1.0f + __expf(-(red[0] + b_copy[0])));
}

// ---------------------------------------------------------------------------
// Fragment load per ISA 7.12.2 (16-bit A/B, K-major source):
// lane L, row=L&15, half=L>>4:
//   VGPR0..3 hold K = half*8 .. +7   (16B)
//   VGPR4..7 hold K = 16+half*8 ..+7 (16B, +32 bytes from lo)
// ---------------------------------------------------------------------------
__device__ __forceinline__ v16bf load_frag(const unsigned short* __restrict__ p) {
  v8bf lo = *(const v8bf*)(p);
  v8bf hi = *(const v8bf*)(p + 16);
  v16bf r;
#pragma unroll
  for (int i = 0; i < 8; ++i) { r[i] = lo[i]; r[i + 8] = hi[i]; }
  return r;
}

// ---------------------------------------------------------------------------
// WMMA GEMM: logits[BT,VV] = Hbf[BT,DD] @ Wbf[VV,DD]^T + bias -> out[:,0:VV]
// Block tile 64(M) x 128(N), 8 waves, each wave 32x32 via 2x2 16x16 subtiles.
// ---------------------------------------------------------------------------
#if USE_ASYNC_LDS

#define LDA 72   // LDS row stride in u16 elems (144 B) for A chunk 64x64
#define LDB 72   // LDS row stride in u16 elems (144 B) for B chunk 128x64

__global__ __launch_bounds__(256) void gemm_wmma_kernel(const unsigned short* __restrict__ Hbf,
                                                        const unsigned short* __restrict__ Wbf,
                                                        const float* __restrict__ bias,
                                                        float* __restrict__ out) {
  __shared__ unsigned short sA[64 * LDA];    // 9216 B
  __shared__ unsigned short sB[128 * LDB];   // 18432 B

  const int tid  = threadIdx.x;
  const int lane = tid & 31;
  const int wid  = tid >> 5;           // 0..7
  const int wm   = wid >> 2;           // 0..1
  const int wn   = wid & 3;            // 0..3
  const int tileM = blockIdx.y * 64;
  const int tileN = blockIdx.x * 128;

  // Async-copy assignments. A chunk: 64 rows x 128B (8x16B pieces/row);
  // thread (r = tid>>2, c = tid&3) copies pieces c and c+4 (64B apart in
  // BOTH global and LDS, so a single immediate offset covers the pair).
  const int ar = tid >> 2, ac = tid & 3;
  const unsigned short* gA = Hbf + (size_t)(tileM + ar) * DD + ac * 8;
  unsigned short* lA = sA + ar * LDA + ac * 8;
  // B chunk: 128 rows x 128B; thread (r = tid>>1) copies 4 consecutive 16B
  // pieces starting at (tid&1)*64B, via immediate offsets 0/16/32/48.
  const int br = tid >> 1, bc = (tid & 1) * 4;
  const unsigned short* gB = Wbf + (size_t)(tileN + br) * DD + bc * 8;
  unsigned short* lB = sB + br * LDB + bc * 8;

  // Per-wave fragment bases in LDS
  const int lrow = lane & 15;
  const int koff = (lane >> 4) * 8;
  const unsigned short* fA0 = sA + (wm * 32 + lrow) * LDA + koff;
  const unsigned short* fA1 = fA0 + 16 * LDA;
  const unsigned short* fB0 = sB + (wn * 32 + lrow) * LDB + koff;
  const unsigned short* fB1 = fB0 + 16 * LDB;

  v8f c00 = {}, c01 = {}, c10 = {}, c11 = {};

  for (int kc = 0; kc < DD; kc += 64) {
    __builtin_amdgcn_global_load_async_to_lds_b128((gptr_b128)(gA + kc), (lptr_b128)lA, 0, 0);
    __builtin_amdgcn_global_load_async_to_lds_b128((gptr_b128)(gA + kc), (lptr_b128)lA, 64, 0);
    __builtin_amdgcn_global_load_async_to_lds_b128((gptr_b128)(gB + kc), (lptr_b128)lB, 0, 0);
    __builtin_amdgcn_global_load_async_to_lds_b128((gptr_b128)(gB + kc), (lptr_b128)lB, 16, 0);
    __builtin_amdgcn_global_load_async_to_lds_b128((gptr_b128)(gB + kc), (lptr_b128)lB, 32, 0);
    __builtin_amdgcn_global_load_async_to_lds_b128((gptr_b128)(gB + kc), (lptr_b128)lB, 48, 0);
#if __has_builtin(__builtin_amdgcn_s_wait_asynccnt)
    __builtin_amdgcn_s_wait_asynccnt(0);
#else
    asm volatile("s_wait_asynccnt 0x0" ::: "memory");
#endif
    __syncthreads();

#pragma unroll
    for (int ks = 0; ks < 64; ks += 32) {
      v16bf a0 = load_frag(fA0 + ks);
      v16bf a1 = load_frag(fA1 + ks);
      v16bf b0 = load_frag(fB0 + ks);
      v16bf b1 = load_frag(fB1 + ks);
      c00 = __builtin_amdgcn_wmma_f32_16x16x32_bf16(false, a0, false, b0, (short)0, c00, false, false);
      c01 = __builtin_amdgcn_wmma_f32_16x16x32_bf16(false, a0, false, b1, (short)0, c01, false, false);
      c10 = __builtin_amdgcn_wmma_f32_16x16x32_bf16(false, a1, false, b0, (short)0, c10, false, false);
      c11 = __builtin_amdgcn_wmma_f32_16x16x32_bf16(false, a1, false, b1, (short)0, c11, false, false);
    }
    __syncthreads();  // before next chunk overwrites LDS
  }

  // C/D layout: lane L -> col n = L&15; VGPR r -> row m = r + 8*(L>>4)
  const int n0 = tileN + wn * 32 + lrow;
  const float bias0 = bias[n0];
  const float bias1 = bias[n0 + 16];
  const int mbase = tileM + wm * 32 + 8 * (lane >> 4);
#pragma unroll
  for (int r = 0; r < 8; ++r) {
    const size_t m0 = (size_t)(mbase + r) * LDOUT;
    const size_t m1 = (size_t)(mbase + r + 16) * LDOUT;
    out[m0 + n0]      = c00[r] + bias0;
    out[m0 + n0 + 16] = c01[r] + bias1;
    out[m1 + n0]      = c10[r] + bias0;
    out[m1 + n0 + 16] = c11[r] + bias1;
  }
}

#else  // fallback: direct-global fragment loads

__global__ __launch_bounds__(256) void gemm_wmma_kernel(const unsigned short* __restrict__ Hbf,
                                                        const unsigned short* __restrict__ Wbf,
                                                        const float* __restrict__ bias,
                                                        float* __restrict__ out) {
  const int lane = threadIdx.x & 31;
  const int wid  = threadIdx.x >> 5;
  const int wm   = wid >> 2;
  const int wn   = wid & 3;

  const int tileM = blockIdx.y * 64  + wm * 32;
  const int tileN = blockIdx.x * 128 + wn * 32;

  const int lrow = lane & 15;
  const int koff = (lane >> 4) * 8;

  const unsigned short* A0 = Hbf + (size_t)(tileM + lrow)      * DD + koff;
  const unsigned short* A1 = Hbf + (size_t)(tileM + lrow + 16) * DD + koff;
  const unsigned short* B0 = Wbf + (size_t)(tileN + lrow)      * DD + koff;
  const unsigned short* B1 = Wbf + (size_t)(tileN + lrow + 16) * DD + koff;

  v8f c00 = {}, c01 = {}, c10 = {}, c11 = {};

#pragma unroll 4
  for (int k = 0; k < DD; k += 32) {
    v16bf a0 = load_frag(A0 + k);
    v16bf a1 = load_frag(A1 + k);
    v16bf b0 = load_frag(B0 + k);
    v16bf b1 = load_frag(B1 + k);
    c00 = __builtin_amdgcn_wmma_f32_16x16x32_bf16(false, a0, false, b0, (short)0, c00, false, false);
    c01 = __builtin_amdgcn_wmma_f32_16x16x32_bf16(false, a0, false, b1, (short)0, c01, false, false);
    c10 = __builtin_amdgcn_wmma_f32_16x16x32_bf16(false, a1, false, b0, (short)0, c10, false, false);
    c11 = __builtin_amdgcn_wmma_f32_16x16x32_bf16(false, a1, false, b1, (short)0, c11, false, false);
  }

  const int n0 = tileN + lrow;
  const float bias0 = bias[n0];
  const float bias1 = bias[n0 + 16];
  const int mbase = tileM + 8 * (lane >> 4);
#pragma unroll
  for (int r = 0; r < 8; ++r) {
    const size_t m0 = (size_t)(mbase + r) * LDOUT;
    const size_t m1 = (size_t)(mbase + r + 16) * LDOUT;
    out[m0 + n0]      = c00[r] + bias0;
    out[m0 + n0 + 16] = c01[r] + bias1;
    out[m1 + n0]      = c10[r] + bias0;
    out[m1 + n0 + 16] = c11[r] + bias1;
  }
}

#endif  // USE_ASYNC_LDS

// ---------------------------------------------------------------------------
// Row softmax over logits (cols [0,VV)), PAD masked, scaled by (1 - gate),
// normalized in place. One 512-thread block per row; online (m,s) reduction.
// ---------------------------------------------------------------------------
__global__ __launch_bounds__(512) void softmax_kernel(float* __restrict__ out,
                                                      const float* __restrict__ gate) {
  const int row = blockIdx.x;
  const int tid = threadIdx.x;
  float* p = out + (size_t)row * LDOUT;

  float m = -__builtin_inff();
  float s = 0.0f;
  for (int c = tid; c < VV; c += 512) {
    if (c == PAD) continue;
    float x = p[c];
    if (x > m) { s = s * __expf(m - x) + 1.0f; m = x; }
    else       { s += __expf(x - m); }
  }

  __shared__ float sm[512];
  __shared__ float ss[512];
  sm[tid] = m; ss[tid] = s;
  __syncthreads();
  for (int off = 256; off > 0; off >>= 1) {
    if (tid < off) {
      float ma = sm[tid], mb = sm[tid + off];
      float M  = fmaxf(ma, mb);
      ss[tid] = ss[tid] * __expf(ma - M) + ss[tid + off] * __expf(mb - M);
      sm[tid] = M;
    }
    __syncthreads();
  }
  const float M   = sm[0];
  const float inv = (1.0f - gate[row]) / ss[0];

  for (int c = tid; c < VV; c += 512) {
    p[c] = (c == PAD) ? 0.0f : __expf(p[c] - M) * inv;
  }
}

// ---------------------------------------------------------------------------
// Copy branch: out[row, VV+cv] = gate[row] * sum_s attn[row,s]*src_map[b,s,cv]
// ---------------------------------------------------------------------------
__global__ __launch_bounds__(128) void copyprob_kernel(const float* __restrict__ attn,
                                                       const float* __restrict__ src_map,
                                                       const float* __restrict__ gate,
                                                       float* __restrict__ out) {
  const int row = blockIdx.y;
  const int cv  = blockIdx.x * 128 + threadIdx.x;
  if (cv >= CVV) return;
  const int b = row / TT;
  const float* a = attn + (size_t)row * SS;
  const float* mp = src_map + (size_t)b * SS * CVV + cv;
  float acc = 0.0f;
#pragma unroll 4
  for (int s = 0; s < SS; ++s) {
    acc = fmaf(a[s], mp[(size_t)s * CVV], acc);
  }
  out[(size_t)row * LDOUT + VV + cv] = acc * gate[row];
}

// ---------------------------------------------------------------------------
// Launch
// ---------------------------------------------------------------------------
extern "C" void kernel_launch(void* const* d_in, const int* in_sizes, int n_in,
                              void* d_out, int out_size, void* d_ws, size_t ws_size,
                              hipStream_t stream) {
  (void)in_sizes; (void)n_in; (void)out_size; (void)ws_size;

  const float* hidden   = (const float*)d_in[0];  // [BT, DD]
  const float* copyattn = (const float*)d_in[1];  // [BT, SS]
  const float* src_map  = (const float*)d_in[2];  // [BB, SS, CVV]
  const float* W        = (const float*)d_in[3];  // [VV, DD]
  const float* bias     = (const float*)d_in[4];  // [VV]
  const float* w_copy   = (const float*)d_in[5];  // [DD]
  const float* b_copy   = (const float*)d_in[6];  // [1]
  float* out = (float*)d_out;                     // [BT, LDOUT]

  // Workspace layout
  char* ws = (char*)d_ws;
  unsigned short* Hbf = (unsigned short*)ws;                              // 4 MB
  unsigned short* Wbf = (unsigned short*)(ws + (size_t)BT * DD * 2);      // 32 MB
  float* gate = (float*)(ws + (size_t)BT * DD * 2 + (size_t)VV * DD * 2); // 16 KB

  // 1) bf16 conversions
  {
    int nH = BT * DD;
    cvt_bf16_kernel<<<(nH + 255) / 256, 256, 0, stream>>>(hidden, Hbf, nH);
    int nW = VV * DD;
    cvt_bf16_kernel<<<(nW + 255) / 256, 256, 0, stream>>>(W, Wbf, nW);
  }

  // 2) copy gate
  gate_kernel<<<BT, 256, 0, stream>>>(hidden, w_copy, b_copy, gate);

  // 3) WMMA GEMM -> biased logits into out[:, 0:VV]
  {
    dim3 grid(VV / 128, BT / 64);  // 250 x 64
    gemm_wmma_kernel<<<grid, 256, 0, stream>>>(Hbf, Wbf, bias, out);
  }

  // 4) softmax (masked, scaled by 1-gate), in place
  softmax_kernel<<<BT, 512, 0, stream>>>(out, gate);

  // 5) copy branch -> out[:, VV:VV+CVV]
  {
    dim3 grid((CVV + 127) / 128, BT);  // 5 x 4096
    copyprob_kernel<<<grid, 128, 0, stream>>>(copyattn, src_map, gate, out);
  }
}
